// NomicMoE_14173392077013
// MI455X (gfx1250) — matrix operand order repaired and checked by reading, not compiled, not run on hardware
//
#include <hip/hip_runtime.h>
#include <hip/hip_bf16.h>
#include <math.h>

#define NE 8
#define NH 768
#define NI 3072
#define NT 8192
#define TILE 32        // tokens per block (two 16-row WMMA M-tiles -> 2x B reuse)
#define ICH 1536       // intermediate-dim chunk held in LDS
#define XS  776        // xs row stride (bf16): 1552B = 388 dw, %64 = 4 -> conflict-free
#define HS2 1544       // hs row stride (bf16): 3088B = 772 dw, %64 = 4
#define NJ2 (NH / 16 / 8)   // GEMM2 n-tiles per wave = 6

typedef __bf16 v16bf __attribute__((ext_vector_type(16)));
typedef float  v8f   __attribute__((ext_vector_type(8)));

union Frag { v16bf v; unsigned u[8]; uint4 q[2]; };

// (a>>16) | (b & 0xFFFF0000) in one v_perm_b32: bytes {b3,b2,a3,a2}
__device__ __forceinline__ unsigned pk_bf16(float a, float b) {
    return __builtin_amdgcn_perm(__float_as_uint(b), __float_as_uint(a), 0x07060302u);
}
__device__ __forceinline__ unsigned short bf16_1(float a) {
    return (unsigned short)(__float_as_uint(a) >> 16);
}
__device__ __forceinline__ void load_afrag(Frag& a, const unsigned short* row, int kA) {
    a.q[0] = *(const uint4*)(row + kA + (kA & 8));         // ds_load_b128
    a.q[1] = *(const uint4*)(row + kA + (kA & 8) + 8);    // ds_load_b128
}
template <bool PRE>
__device__ __forceinline__ void load_bfrag(Frag& b, const unsigned short* wrowb,
                                           const float* wrowf, int kB) {
    if (PRE) {
        b.q[0] = *(const uint4*)(wrowb + kB);       // global_load_b128
        b.q[1] = *(const uint4*)(wrowb + kB + 8);   // global_load_b128
    } else {
#pragma unroll
        for (int i2 = 0; i2 < 8; i2++) {
            float2 f = *(const float2*)(wrowf + kB + 2 * i2);
            b.u[i2] = pk_bf16(f.x, f.y);
        }
    }
}

// ---------------- Kernel 0: bulk fp32 -> bf16 convert -------------------------------
__global__ void cvt_bf16_kernel(const float4* __restrict__ src, uint4* __restrict__ dst,
                                long n8) {
    long i = (long)blockIdx.x * blockDim.x + threadIdx.x;
    const long stride = (long)gridDim.x * blockDim.x;
    for (; i < n8; i += stride) {
        float4 a = src[2 * i], b = src[2 * i + 1];
        uint4 o;
        o.x = pk_bf16(a.x, a.y); o.y = pk_bf16(a.z, a.w);
        o.z = pk_bf16(b.x, b.y); o.w = pk_bf16(b.z, b.w);
        dst[i] = o;
    }
}

// ---------------- Kernel 1: init output with bias, zero expert counters -------------
__global__ void moe_init_kernel(float* __restrict__ out, const float* __restrict__ bias,
                                int* __restrict__ cnt) {
    size_t i = (size_t)blockIdx.x * blockDim.x + threadIdx.x;
    if (i < NE) cnt[i] = 0;
    const size_t total = (size_t)NT * NH;
    const size_t stride = (size_t)gridDim.x * blockDim.x;
    for (size_t idx = i; idx < total; idx += stride)
        out[idx] = bias[idx % NH];
}

// ---------------- Kernel 2: router (one wave32 per token) ---------------------------
__global__ __launch_bounds__(256) void moe_router_kernel(
    const float* __restrict__ x, const float* __restrict__ rweight,
    int* __restrict__ cnt, int* __restrict__ lists, float* __restrict__ glist) {
    __shared__ float rw[NE * NH];
    for (int i = threadIdx.x; i < NE * NH; i += 256) rw[i] = rweight[i];
    __syncthreads();

    const int lane = threadIdx.x & 31;
    const int t = (blockIdx.x * 256 + threadIdx.x) >> 5;
    const float* xr = x + (size_t)t * NH;

    float s[NE];
#pragma unroll
    for (int e = 0; e < NE; e++) s[e] = 0.f;
    for (int k = lane; k < NH; k += 32) {
        float xv = xr[k];
#pragma unroll
        for (int e = 0; e < NE; e++) s[e] += xv * rw[e * NH + k];
    }
#pragma unroll
    for (int off = 16; off > 0; off >>= 1) {
#pragma unroll
        for (int e = 0; e < NE; e++) s[e] += __shfl_xor(s[e], off, 32);
    }
    if (lane == 0) {
        float mx = s[0];
#pragma unroll
        for (int e = 1; e < NE; e++) mx = fmaxf(mx, s[e]);
        float p[NE]; float sum = 0.f;
#pragma unroll
        for (int e = 0; e < NE; e++) { p[e] = __expf(s[e] - mx); sum += p[e]; }
        const float inv = 1.f / sum;
        int e0 = 0; float p0 = p[0];
#pragma unroll
        for (int e = 1; e < NE; e++) if (p[e] > p0) { p0 = p[e]; e0 = e; }
        int e1 = -1; float p1 = -1.f;
#pragma unroll
        for (int e = 0; e < NE; e++) if (e != e0 && p[e] > p1) { p1 = p[e]; e1 = e; }
        int s0 = atomicAdd(&cnt[e0], 1);
        lists[e0 * NT + s0] = t;  glist[e0 * NT + s0] = p0 * inv;
        int s1 = atomicAdd(&cnt[e1], 1);
        lists[e1 * NT + s1] = t;  glist[e1 * NT + s1] = p1 * inv;
    }
}

// ---------------- Kernel 3: fused expert MLP, 32-token tiles, I-chunked -------------
template <bool PRE>
__global__ __launch_bounds__(256) void moe_mlp_kernel(
    const void* __restrict__ xp, const void* __restrict__ w1p,
    const void* __restrict__ w2p, float* __restrict__ out,
    const int* __restrict__ cnt, const int* __restrict__ lists,
    const float* __restrict__ glist) {
    __shared__ unsigned short xs[TILE * XS];    // 48.5 KB  x-tile bf16
    __shared__ unsigned short hs[TILE * HS2];   // 96.5 KB  h-chunk bf16 (32 x 1536)
    __shared__ int   tsh[TILE];
    __shared__ float gsh[TILE];

    const int e = blockIdx.y;
    const int tile = blockIdx.x;
    const int count = cnt[e];
    if (tile * TILE >= count) return;

    const int tid = threadIdx.x;
    if (tid < TILE) {
        int r = tile * TILE + tid;
        if (r < count) { tsh[tid] = lists[e * NT + r]; gsh[tid] = glist[e * NT + r]; }
        else           { tsh[tid] = 0;                 gsh[tid] = 0.f; }
    }
    __syncthreads();

    // Stage 32 gathered token rows into LDS as bf16
    if (PRE) {
        const unsigned* xb = (const unsigned*)xp;
        for (int idx = tid; idx < TILE * (NH / 2); idx += 256) {
            int r = idx / (NH / 2), c = idx - r * (NH / 2);
            *(unsigned*)&xs[r * XS + 2 * c] = xb[(size_t)tsh[r] * (NH / 2) + c];
        }
    } else {
        const float* x = (const float*)xp;
        for (int idx = tid; idx < TILE * (NH / 2); idx += 256) {
            int r = idx / (NH / 2), c = idx - r * (NH / 2);
            float2 f = *(const float2*)(x + (size_t)tsh[r] * NH + 2 * c);
            *(unsigned*)&xs[r * XS + 2 * c] = pk_bf16(f.x, f.y);
        }
    }
    __syncthreads();

    const int lane = tid & 31;
    const int wv   = tid >> 5;
    const int ln   = lane & 15;
    const int hi   = lane >> 4;

    // Persistent GEMM2 accumulators: 6 n-tiles x 2 m-tiles (compile-time indexed)
    v8f acc2[NJ2 * 2];
#pragma unroll
    for (int i = 0; i < NJ2 * 2; i++) acc2[i] = (v8f){};

    for (int c = 0; c < NI / ICH; c++) {
        if (c) __syncthreads();    // GEMM2 reads of previous chunk done before rewrite

        // ---- GEMM1 chunk: hs[32 x ICH] = gelu(x @ w1[c-range]^T) ----
        for (int j = 0; j < ICH / 16 / 8; j++) {       // 12 n-tiles per wave
            const int ntl = wv + 8 * j;
            const int ng  = c * ICH + ntl * 16;        // global intermediate base
            const unsigned short* x0 = &xs[ln * XS];
            const unsigned short* x1 = &xs[(16 + ln) * XS];
            const float*          wrowf = PRE ? nullptr
                : (const float*)w1p + ((size_t)e * NI + ng + ln) * NH;
            const unsigned short* wrowb = PRE
                ? (const unsigned short*)w1p + ((size_t)e * NI + ng + ln) * NH : nullptr;
            v8f a0 = {}, a1 = {};
            Frag b[2], f0[2], f1[2];
            load_bfrag<PRE>(b[0], wrowb, wrowf, hi * 16);
            load_afrag(f0[0], x0, hi * 8);
            load_afrag(f1[0], x1, hi * 8);
            constexpr int NP1 = NH / 64;               // 12 pipelined pairs
#pragma unroll 1
            for (int s2 = 0; s2 < NP1; s2++) {
                const int s = 2 * s2;
                load_bfrag<PRE>(b[1], wrowb, wrowf, (s + 1) * 32 + hi * 16);
                load_afrag(f0[1], x0, (s + 1) * 32 + hi * 8);
                load_afrag(f1[1], x1, (s + 1) * 32 + hi * 8);
                a0 = __builtin_amdgcn_wmma_f32_16x16x32_bf16(
                    false, f0[0].v, false, b[0].v, (short)0, a0, false, false);
                a1 = __builtin_amdgcn_wmma_f32_16x16x32_bf16(
                    false, f1[0].v, false, b[0].v, (short)0, a1, false, false);
                if (s2 + 1 < NP1) {
                    load_bfrag<PRE>(b[0], wrowb, wrowf, (s + 2) * 32 + hi * 16);
                    load_afrag(f0[0], x0, (s + 2) * 32 + hi * 8);
                    load_afrag(f1[0], x1, (s + 2) * 32 + hi * 8);
                }
                a0 = __builtin_amdgcn_wmma_f32_16x16x32_bf16(
                    false, f0[1].v, false, b[1].v, (short)0, a0, false, false);
                a1 = __builtin_amdgcn_wmma_f32_16x16x32_bf16(
                    false, f1[1].v, false, b[1].v, (short)0, a1, false, false);
            }
#pragma unroll
            for (int i2 = 0; i2 < 8; i2++) {
                float v0 = a0[i2], v1 = a1[i2];
                float g0 = 0.5f * v0 * (1.0f + erff(v0 * 0.70710678118654752f));
                float g1 = 0.5f * v1 * (1.0f + erff(v1 * 0.70710678118654752f));
                hs[(i2 + hi * 8) * HS2 + ntl * 16 + ln]      = bf16_1(g0);
                hs[(16 + i2 + hi * 8) * HS2 + ntl * 16 + ln] = bf16_1(g1);
            }
        }
        __syncthreads();

        // ---- GEMM2 partial: acc2 += h_chunk @ w2[:, c-range]^T ----
#pragma unroll
        for (int j = 0; j < NJ2; j++) {
            const int nb = (wv + 8 * j) * 16;
            const unsigned short* h0 = &hs[ln * HS2];
            const unsigned short* h1 = &hs[(16 + ln) * HS2];
            const float*          wrowf = PRE ? nullptr
                : (const float*)w2p + ((size_t)e * NH + nb + ln) * NI + c * ICH;
            const unsigned short* wrowb = PRE
                ? (const unsigned short*)w2p + ((size_t)e * NH + nb + ln) * NI + c * ICH
                : nullptr;
            Frag b[2], f0[2], f1[2];
            load_bfrag<PRE>(b[0], wrowb, wrowf, hi * 16);
            load_afrag(f0[0], h0, hi * 8);
            load_afrag(f1[0], h1, hi * 8);
            constexpr int NP2 = ICH / 64;              // 24 pipelined pairs
#pragma unroll 1
            for (int s2 = 0; s2 < NP2; s2++) {
                const int s = 2 * s2;
                load_bfrag<PRE>(b[1], wrowb, wrowf, (s + 1) * 32 + hi * 16);
                load_afrag(f0[1], h0, (s + 1) * 32 + hi * 8);
                load_afrag(f1[1], h1, (s + 1) * 32 + hi * 8);
                acc2[2 * j] = __builtin_amdgcn_wmma_f32_16x16x32_bf16(
                    false, f0[0].v, false, b[0].v, (short)0, acc2[2 * j], false, false);
                acc2[2 * j + 1] = __builtin_amdgcn_wmma_f32_16x16x32_bf16(
                    false, f1[0].v, false, b[0].v, (short)0, acc2[2 * j + 1], false, false);
                if (s2 + 1 < NP2) {
                    load_bfrag<PRE>(b[0], wrowb, wrowf, (s + 2) * 32 + hi * 16);
                    load_afrag(f0[0], h0, (s + 2) * 32 + hi * 8);
                    load_afrag(f1[0], h1, (s + 2) * 32 + hi * 8);
                }
                acc2[2 * j] = __builtin_amdgcn_wmma_f32_16x16x32_bf16(
                    false, f0[1].v, false, b[1].v, (short)0, acc2[2 * j], false, false);
                acc2[2 * j + 1] = __builtin_amdgcn_wmma_f32_16x16x32_bf16(
                    false, f1[1].v, false, b[1].v, (short)0, acc2[2 * j + 1], false, false);
            }
        }
    }

    // ---- Epilogue: gate-scale and accumulate into output ----
#pragma unroll
    for (int j = 0; j < NJ2; j++) {
        const int nb = (wv + 8 * j) * 16;
#pragma unroll
        for (int mt = 0; mt < 2; mt++) {
#pragma unroll
            for (int i2 = 0; i2 < 8; i2++) {
                const int md = mt * 16 + i2 + 8 * hi;
                float val = gsh[md] * acc2[2 * j + mt][i2];
                unsafeAtomicAdd(out + (size_t)tsh[md] * NH + nb + ln, val);
            }
        }
    }
}

// ---------------- Launcher ----------------------------------------------------------
extern "C" void kernel_launch(void* const* d_in, const int* in_sizes, int n_in,
                              void* d_out, int out_size, void* d_ws, size_t ws_size,
                              hipStream_t stream) {
    const float* x    = (const float*)d_in[0];  // [T, H]
    const float* rw   = (const float*)d_in[1];  // [E, H]
    const float* w1   = (const float*)d_in[2];  // [E, I, H]
    const float* w2   = (const float*)d_in[3];  // [E, H, I]
    const float* bias = (const float*)d_in[4];  // [H]
    float* out = (float*)d_out;                 // [T, H]

    const size_t OFF_LISTS = 256;
    const size_t OFF_GLIST = OFF_LISTS + (size_t)NE * NT * sizeof(int);
    const size_t OFF_XB    = OFF_GLIST + (size_t)NE * NT * sizeof(float);
    const size_t N_X  = (size_t)NT * NH;
    const size_t N_W  = (size_t)NE * NI * NH;
    const size_t OFF_W1B = OFF_XB + N_X * 2;
    const size_t OFF_W2B = OFF_W1B + N_W * 2;
    const size_t NEED    = OFF_W2B + N_W * 2;   // ~88.6 MB

    int*   cnt   = (int*)d_ws;
    int*   lists = (int*)((char*)d_ws + OFF_LISTS);
    float* glist = (float*)((char*)d_ws + OFF_GLIST);

    moe_init_kernel<<<1024, 256, 0, stream>>>(out, bias, cnt);
    moe_router_kernel<<<NT / 8, 256, 0, stream>>>(x, rw, cnt, lists, glist);

    if (ws_size >= NEED) {
        uint4* xb  = (uint4*)((char*)d_ws + OFF_XB);
        uint4* w1b = (uint4*)((char*)d_ws + OFF_W1B);
        uint4* w2b = (uint4*)((char*)d_ws + OFF_W2B);
        cvt_bf16_kernel<<<512,  256, 0, stream>>>((const float4*)x,  xb,  (long)(N_X / 8));
        cvt_bf16_kernel<<<2048, 256, 0, stream>>>((const float4*)w1, w1b, (long)(N_W / 8));
        cvt_bf16_kernel<<<2048, 256, 0, stream>>>((const float4*)w2, w2b, (long)(N_W / 8));
        moe_mlp_kernel<true><<<dim3(NT / TILE, NE), 256, 0, stream>>>(
            (const void*)xb, (const void*)w1b, (const void*)w2b, out, cnt, lists, glist);
    } else {
        moe_mlp_kernel<false><<<dim3(NT / TILE, NE), 256, 0, stream>>>(
            (const void*)x, (const void*)w1, (const void*)w2, out, cnt, lists, glist);
    }
}
